// NewMinibatchEnergyDistance_68865505624111
// MI455X (gfx1250) — matrix operand order repaired and checked by reading, not compiled, not run on hardware
//
#include <hip/hip_runtime.h>

typedef __attribute__((ext_vector_type(16))) __bf16 v16bf;
typedef __attribute__((ext_vector_type(8)))  __bf16 v8bf;
typedef __attribute__((ext_vector_type(4)))  __bf16 v4bf;
typedef __attribute__((ext_vector_type(8)))  float  v8f;
typedef __attribute__((ext_vector_type(4)))  float  v4f;

#define B_      200
#define M_TOT   800
#define M_PADR  896     /* 56 M-tiles of 16: 7 per wave x 8 waves, no WMMA predication */
#define D_IN_   3072
#define D_F_    16384
#define EPS_INV 10.0f   /* 1/0.1 */
#define BN_     64      /* N columns per block in the feature GEMM */

__device__ __forceinline__ __bf16 f2bf(float x) { return (__bf16)x; }

__device__ __forceinline__ void pair_ab(int p, int& a, int& b) {
    // reference pair order: (x,x'),(x,y),(x,y'),(x',y),(x',y'),(y,y')
    if      (p == 0) { a = 0; b = 1; }
    else if (p == 1) { a = 0; b = 2; }
    else if (p == 2) { a = 0; b = 3; }
    else if (p == 3) { a = 1; b = 2; }
    else if (p == 4) { a = 1; b = 3; }
    else             { a = 2; b = 3; }
}

// -------------------------------------------------------------------------
// Kernel 0: pack [x;x';y;y'] into bf16 Abf (896x3072), rows >= 800 zeroed.
// 5.5 MB -> L2-resident for the whole GEMM; removes per-k-step LDS staging.
// -------------------------------------------------------------------------
__global__ __launch_bounds__(256) void k_pack_a(
    const float* __restrict__ x,  const float* __restrict__ xp,
    const float* __restrict__ y,  const float* __restrict__ yp,
    unsigned short* __restrict__ Aq)
{
    __bf16* Abf = reinterpret_cast<__bf16*>(Aq);
    int i4 = blockIdx.x * 256 + threadIdx.x;               // float4 index
    if (i4 >= M_PADR * (D_IN_ / 4)) return;
    int m = i4 / (D_IN_ / 4);
    int c = (i4 - m * (D_IN_ / 4)) << 2;
    v4f val;
    if (m < M_TOT) {
        int g = m / B_;
        int r = m - g * B_;
        const float* p = (g == 0) ? x : (g == 1) ? xp : (g == 2) ? y : yp;
        val = *(const v4f*)(p + (size_t)r * D_IN_ + c);
    } else {
        val[0] = val[1] = val[2] = val[3] = 0.f;
    }
    v4bf o;
    o[0] = f2bf(val[0]); o[1] = f2bf(val[1]); o[2] = f2bf(val[2]); o[3] = f2bf(val[3]);
    *(v4bf*)(Abf + (size_t)m * D_IN_ + c) = o;
}

// -------------------------------------------------------------------------
// Kernel 1: G = Abf @ W  (800x3072 @ 3072x16384), bf16 WMMA, f32 acc.
// Grid over N only (256 blocks x 64 cols) => W streams from HBM once.
// 8 waves; each wave: 7 M-tiles x 4 N-tiles = 28 accumulators.
// A frags: direct 16B loads from L2-resident Abf. W tile: LDS (fp32->bf16,
// transposed) shared by all waves.
// -------------------------------------------------------------------------
__global__ __launch_bounds__(256) void k_feat_gemm(
    const unsigned short* __restrict__ Aq,
    const float* __restrict__ W, unsigned short* __restrict__ Gq)
{
    const __bf16* Abf = reinterpret_cast<const __bf16*>(Aq);
    __bf16* G = reinterpret_cast<__bf16*>(Gq);
    __shared__ __align__(32) __bf16 sB[BN_ * 32];          // 4 KB, sB[n*32+k]

    const int tid  = threadIdx.x;
    const int lane = tid & 31;
    const int wv   = tid >> 5;
    const int lm   = lane & 15;
    const int hf   = lane >> 4;
    const int n0   = blockIdx.x * BN_;

    v8f acc[7][4];
#pragma unroll
    for (int i = 0; i < 7; ++i)
#pragma unroll
        for (int nn = 0; nn < 4; ++nn)
#pragma unroll
            for (int t = 0; t < 8; ++t) acc[i][nn][t] = 0.f;

    for (int k0 = 0; k0 < D_IN_; k0 += 32) {
        // ---- W tile (32 x 64) -> LDS bf16 transposed (coalesced over n)
        for (int e = tid; e < 32 * BN_; e += 256) {
            int k = e >> 6;           // e / 64
            int n = e & (BN_ - 1);
            sB[n * 32 + k] = f2bf(W[(size_t)(k0 + k) * D_F_ + n0 + n]);
        }
        __syncthreads();

        v16bf bfr[4];
#pragma unroll
        for (int nn = 0; nn < 4; ++nn)
            bfr[nn] = *(const v16bf*)&sB[(nn * 16 + lm) * 32 + hf * 16];

#pragma unroll
        for (int i = 0; i < 7; ++i) {
            int m = (wv + i * 8) * 16 + lm;
            const __bf16* arow = Abf + (size_t)m * D_IN_ + k0 + hf * 16;
            v8bf a0 = *(const v8bf*)(arow);
            v8bf a1 = *(const v8bf*)(arow + 8);
            v16bf af;
#pragma unroll
            for (int t = 0; t < 8; ++t) { af[t] = a0[t]; af[t + 8] = a1[t]; }
#pragma unroll
            for (int nn = 0; nn < 4; ++nn)
                acc[i][nn] = __builtin_amdgcn_wmma_f32_16x16x32_bf16(
                    false, af, false, bfr[nn], (short)0, acc[i][nn], false, false);
        }
        __syncthreads();
    }

#pragma unroll
    for (int i = 0; i < 7; ++i) {
        int mt = wv + i * 8;
        if (mt < 50) {
#pragma unroll
            for (int nn = 0; nn < 4; ++nn)
#pragma unroll
                for (int r = 0; r < 8; ++r) {
                    int m = mt * 16 + r + hf * 8;
                    G[(size_t)m * D_F_ + n0 + nn * 16 + lm] = f2bf(acc[i][nn][r]);
                }
        }
    }
}

// -------------------------------------------------------------------------
// Kernel 2: S = G @ G^T (800x800, K=16384). G bf16, L2-resident.
// One wave per 32x32 output region (2x2 tiles of 16x16, A/B frag reuse).
// -------------------------------------------------------------------------
__global__ __launch_bounds__(32) void k_gram(
    const unsigned short* __restrict__ Gq, float* __restrict__ S)
{
    const __bf16* G = reinterpret_cast<const __bf16*>(Gq);
    const int lane = threadIdx.x & 31;
    const int lm   = lane & 15;
    const int hf   = lane >> 4;
    const int mt0  = blockIdx.y * 2, nt0 = blockIdx.x * 2;

    v8f acc[2][2];
#pragma unroll
    for (int i = 0; i < 2; ++i)
#pragma unroll
        for (int j = 0; j < 2; ++j)
#pragma unroll
            for (int t = 0; t < 8; ++t) acc[i][j][t] = 0.f;

    const size_t rA0 = (size_t)((mt0 + 0) * 16 + lm) * D_F_;
    const size_t rA1 = (size_t)((mt0 + 1) * 16 + lm) * D_F_;
    const size_t rB0 = (size_t)((nt0 + 0) * 16 + lm) * D_F_;
    const size_t rB1 = (size_t)((nt0 + 1) * 16 + lm) * D_F_;

#pragma unroll 2
    for (int k0 = 0; k0 < D_F_; k0 += 32) {
        v16bf af[2], bf[2];
#pragma unroll
        for (int i = 0; i < 2; ++i) {
            const __bf16* ar = G + (i ? rA1 : rA0) + k0 + hf * 16;
            v8bf a0 = *(const v8bf*)(ar);
            v8bf a1 = *(const v8bf*)(ar + 8);
#pragma unroll
            for (int t = 0; t < 8; ++t) { af[i][t] = a0[t]; af[i][t + 8] = a1[t]; }
        }
        bf[0] = *(const v16bf*)(G + rB0 + k0 + hf * 16);
        bf[1] = *(const v16bf*)(G + rB1 + k0 + hf * 16);
#pragma unroll
        for (int i = 0; i < 2; ++i)
#pragma unroll
            for (int j = 0; j < 2; ++j)
                acc[i][j] = __builtin_amdgcn_wmma_f32_16x16x32_bf16(
                    false, af[i], false, bf[j], (short)0, acc[i][j], false, false);
    }

#pragma unroll
    for (int i = 0; i < 2; ++i)
#pragma unroll
        for (int j = 0; j < 2; ++j)
#pragma unroll
            for (int r = 0; r < 8; ++r)
                S[(size_t)((mt0 + i) * 16 + r + hf * 8) * M_TOT
                  + (nt0 + j) * 16 + lm] = acc[i][j][r];
}

// -------------------------------------------------------------------------
// Kernel 3a: inverse row norms from diag(S)
// -------------------------------------------------------------------------
__global__ __launch_bounds__(256) void k_inorm(
    const float* __restrict__ S, float* __restrict__ inorm)
{
    int i = blockIdx.x * 256 + threadIdx.x;
    if (i < M_TOT) inorm[i] = rsqrtf(fmaxf(S[(size_t)i * M_TOT + i], 1e-30f));
}

// -------------------------------------------------------------------------
// Kernel 3b: 6 cost matrices C = 1 - cos, Gibbs kernel K = exp(-C/eps)
// -------------------------------------------------------------------------
__global__ __launch_bounds__(256) void k_costs(
    const float* __restrict__ S, const float* __restrict__ inorm,
    float* __restrict__ C, float* __restrict__ Km)
{
    int idx = blockIdx.x * 256 + threadIdx.x;
    if (idx >= 6 * B_ * B_) return;
    int p   = idx / (B_ * B_);
    int rem = idx - p * (B_ * B_);
    int i   = rem / B_;
    int j   = rem - i * B_;
    int a, b; pair_ab(p, a, b);
    int gi = a * B_ + i, gj = b * B_ + j;
    float cs = S[(size_t)gi * M_TOT + gj] * inorm[gi] * inorm[gj];
    float c  = 1.0f - cs;
    C[idx]  = c;
    Km[idx] = expf(-c * EPS_INV);
}

// -------------------------------------------------------------------------
// Kernel 4: Sinkhorn — one workgroup per pair, u/v in LDS, K/C from L2.
// -------------------------------------------------------------------------
__global__ __launch_bounds__(256) void k_sinkhorn(
    const float* __restrict__ C, const float* __restrict__ Km,
    const int* __restrict__ nIterPtr, float* __restrict__ losses)
{
    __shared__ float u[B_], v[B_], red[256];
    const int tid = threadIdx.x;
    const int p   = blockIdx.x;
    const float* Kp = Km + (size_t)p * B_ * B_;
    const float* Cp = C  + (size_t)p * B_ * B_;
    const int nIter = *nIterPtr;
    const float ab  = 1.0f / (float)B_;

    for (int i = tid; i < B_; i += 256) { u[i] = 1.f; v[i] = 1.f; }
    __syncthreads();

    for (int it = 0; it < nIter; ++it) {
        for (int i = tid; i < B_; i += 256) {
            float s = 0.f;
            const float* row = Kp + (size_t)i * B_;
            for (int j = 0; j < B_; ++j) s += row[j] * v[j];
            u[i] = ab / s;
        }
        __syncthreads();
        for (int j = tid; j < B_; j += 256) {
            float s = 0.f;
            for (int i = 0; i < B_; ++i) s += Kp[(size_t)i * B_ + j] * u[i];
            v[j] = ab / s;
        }
        __syncthreads();
    }

    float part = 0.f;
    for (int idx = tid; idx < B_ * B_; idx += 256) {
        int i = idx / B_;
        int j = idx - i * B_;
        part += u[i] * Kp[idx] * v[j] * Cp[idx];
    }
    red[tid] = part;
    __syncthreads();
    for (int s = 128; s > 0; s >>= 1) {
        if (tid < s) red[tid] += red[tid + s];
        __syncthreads();
    }
    if (tid == 0) losses[p] = red[0];
}

// -------------------------------------------------------------------------
// Kernel 5: final scalar combine
// -------------------------------------------------------------------------
__global__ void k_combine(const float* __restrict__ L, float* __restrict__ out)
{
    out[0] = L[1] + L[2] + L[3] + L[4] - 2.0f * (L[0] + L[5]);
}

extern "C" void kernel_launch(void* const* d_in, const int* in_sizes, int n_in,
                              void* d_out, int out_size, void* d_ws, size_t ws_size,
                              hipStream_t stream)
{
    const float* x  = (const float*)d_in[0];
    const float* xp = (const float*)d_in[1];
    const float* y  = (const float*)d_in[2];
    const float* yp = (const float*)d_in[3];
    const float* W  = (const float*)d_in[4];
    const int*   nI = (const int*)d_in[5];

    char* w = (char*)d_ws;
    size_t o = 0;
    unsigned short* Aq = (unsigned short*)(w + o); o += (size_t)M_PADR * D_IN_ * 2; //  5.5 MB
    unsigned short* Gq = (unsigned short*)(w + o); o += (size_t)M_TOT * D_F_ * 2;   // 26.2 MB
    float* S      = (float*)(w + o); o += (size_t)M_TOT * M_TOT * 4;                //  2.6 MB
    float* inorm  = (float*)(w + o); o += 4096;
    float* C      = (float*)(w + o); o += (size_t)6 * B_ * B_ * 4;
    float* Km     = (float*)(w + o); o += (size_t)6 * B_ * B_ * 4;
    float* losses = (float*)(w + o); o += 256;
    (void)ws_size; (void)in_sizes; (void)n_in; (void)out_size;

    int packTot = M_PADR * (D_IN_ / 4);
    k_pack_a<<<(packTot + 255) / 256, 256, 0, stream>>>(x, xp, y, yp, Aq);

    k_feat_gemm<<<D_F_ / BN_, 256, 0, stream>>>(Aq, W, Gq);

    dim3 g2(25, 25);
    k_gram<<<g2, 32, 0, stream>>>(Gq, S);

    k_inorm<<<(M_TOT + 255) / 256, 256, 0, stream>>>(S, inorm);

    int tot = 6 * B_ * B_;
    k_costs<<<(tot + 255) / 256, 256, 0, stream>>>(S, inorm, C, Km);

    k_sinkhorn<<<6, 256, 0, stream>>>(C, Km, nI, losses);

    k_combine<<<1, 1, 0, stream>>>(losses, (float*)d_out);
}
